// GNN_81913616269585
// MI455X (gfx1250) — matrix operand order repaired and checked by reading, not compiled, not run on hardware
//
#include <hip/hip_runtime.h>
#include <hip/hip_bf16.h>

typedef __bf16 bf16_t;
typedef __attribute__((ext_vector_type(16))) __bf16 v16bf;
typedef __attribute__((ext_vector_type(8)))  __bf16 v8bf;
typedef __attribute__((ext_vector_type(8)))  float  v8f;

#define NN    1024
#define EE    2048
#define CIN   4
#define HID   192
#define KCH   5
#define NLAY  4
#define T1S   208            // t1 row stride in bf16 elems (16B aligned)
#define KTOT  37056          // 193 * 192
#define JTOT  193
#define JCHUNK 24            // j's per grid.y slice (24*2B = 48B, 16B aligned)
#define KSPLIT 9             // ceil(193/24)
#define BSTR  40             // btile col stride in bf16 elems (80B, 16B aligned)

// ---------------------------------------------------------- async helpers
__device__ __forceinline__ void async_g2l_b128(void* lds, const void* g) {
    unsigned loff = (unsigned)(uintptr_t)lds;   // low 32 bits of flat = LDS offset
    unsigned long long ga = (unsigned long long)(uintptr_t)g;
    asm volatile("global_load_async_to_lds_b128 %0, %1, off"
                 :: "v"(loff), "v"(ga) : "memory");
}
__device__ __forceinline__ void wait_async_le6() {
    asm volatile("s_wait_asynccnt 0x6" ::: "memory");
}
__device__ __forceinline__ void wait_async_le3() {
    asm volatile("s_wait_asynccnt 0x3" ::: "memory");
}
__device__ __forceinline__ void wait_async_0() {
    asm volatile("s_wait_asynccnt 0x0" ::: "memory");
}

// ---------------------------------------------------------------- utilities
__global__ void zero_f32(float* p, int n) {
    int i = blockIdx.x * 256 + threadIdx.x;
    if (i < n) p[i] = 0.0f;
}

__global__ void f32_to_bf16(const float* __restrict__ in, bf16_t* __restrict__ out, int n) {
    int i = blockIdx.x * 256 + threadIdx.x;
    if (i < n) out[i] = (bf16_t)in[i];
}

// ------------------------------------------------------ weight conversions
__global__ void conv_t192(const float* __restrict__ in, bf16_t* __restrict__ out) {
    int idx = blockIdx.x * 256 + threadIdx.x;
    if (idx >= NLAY * HID * HID) return;
    int j = idx % HID;
    int r = idx / HID;
    int o = r % HID;
    int i = r / HID;
    out[idx] = (bf16_t)in[(size_t)i * HID * HID + (size_t)j * HID + o];
}

// W2T[i][o][m]: m<36864 -> nn2_w[i][m/192][(m%192)*192+o]; else nn2_b virtual row
__global__ void conv_w2t(const float* __restrict__ w2, const float* __restrict__ b2,
                         bf16_t* __restrict__ out) {
    int idx = blockIdx.x * 256 + threadIdx.x;   // exact grid: 4*192*37056/256
    int m = idx % KTOT;
    int r = idx / KTOT;
    int o = r % HID;
    int i = r / HID;
    float v;
    if (m < HID * HID) {
        int j  = m / HID;
        int ii = m % HID;
        v = w2[((size_t)i * HID + j) * (size_t)(HID * HID) + (size_t)ii * HID + o];
    } else {
        int ii = m - HID * HID;
        v = b2[(size_t)i * (HID * HID) + (size_t)ii * HID + o];
    }
    out[(size_t)idx] = (bf16_t)v;
}

__global__ void t1_ones(bf16_t* __restrict__ t1) {
    int e = blockIdx.x * 256 + threadIdx.x;
    if (e >= EE) return;
    t1[(size_t)e * T1S + 192] = (bf16_t)1.0f;
    for (int c = 193; c < T1S; ++c) t1[(size_t)e * T1S + c] = (bf16_t)0.0f;
}

// ------------------------------------------------------------ ChebConv path
__global__ void count_deg(const int* __restrict__ src, float* __restrict__ deg) {
    int e = blockIdx.x * 256 + threadIdx.x;
    if (e < EE) atomicAdd(&deg[src[e]], 1.0f);
}

__global__ void make_dinv(float* __restrict__ deg) {
    int n = blockIdx.x * 256 + threadIdx.x;
    if (n >= NN) return;
    float d = deg[n];
    deg[n] = (d > 0.0f) ? (1.0f / sqrtf(fmaxf(d, 1.0f))) : 0.0f;
}

__global__ void make_wnorm(const int* __restrict__ src, const int* __restrict__ dst,
                           const float* __restrict__ dinv, float* __restrict__ wn) {
    int e = blockIdx.x * 256 + threadIdx.x;
    if (e < EE) wn[e] = -dinv[src[e]] * dinv[dst[e]];
}

__global__ void lap_apply(const float* __restrict__ v, const int* __restrict__ src,
                          const int* __restrict__ dst, const float* __restrict__ wn,
                          float* __restrict__ y) {
    int e = blockIdx.x * 256 + threadIdx.x;
    if (e >= EE) return;
    float w = wn[e];
    int s = src[e], d = dst[e];
#pragma unroll
    for (int c = 0; c < CIN; ++c) atomicAdd(&y[d * CIN + c], w * v[s * CIN + c]);
}

__global__ void cheb_comb(const float* __restrict__ tmp, const float* __restrict__ tx0,
                          float* __restrict__ tx2) {
    int i = blockIdx.x * 256 + threadIdx.x;
    if (i < NN * CIN) tx2[i] = 2.0f * tmp[i] - tx0[i];
}

__global__ void cheb_mm(const float* __restrict__ Tx, const float* __restrict__ W,
                        const float* __restrict__ bias, float* __restrict__ out, int init) {
    int idx = blockIdx.x * 256 + threadIdx.x;   // N*HID exact
    int n = idx / HID, o = idx % HID;
    float s = init ? bias[o] : out[idx];
#pragma unroll
    for (int c = 0; c < CIN; ++c) s += Tx[n * CIN + c] * W[c * HID + o];
    out[idx] = s;
}

// ------------------------------------------------------------ edge encoder
__global__ void edge_enc(const float* __restrict__ ea, const float* __restrict__ w,
                         const float* __restrict__ b, bf16_t* __restrict__ out) {
    int idx = blockIdx.x * 256 + threadIdx.x;   // E*HID exact
    int e = idx / HID, o = idx % HID;
    float s = b[o];
#pragma unroll
    for (int c = 0; c < CIN; ++c) s += ea[e * CIN + c] * w[c * HID + o];
    out[idx] = (bf16_t)s;
}

// --------------------------------------------------------------- gather src
__global__ void gather_src(const bf16_t* __restrict__ vbf, const int* __restrict__ src,
                           bf16_t* __restrict__ vs) {
    int idx = blockIdx.x * 256 + threadIdx.x;   // E*24 exact (uint4 chunks)
    int e = idx / 24, c = idx % 24;
    const uint4* s = (const uint4*)(vbf + (size_t)src[e] * HID);
    ((uint4*)vs)[(size_t)e * 24 + c] = s[c];
}

// --------------------------------------------------------- WMMA GEMM: t1
__global__ __launch_bounds__(32) void gemm_t1(const bf16_t* __restrict__ A,
                                              const bf16_t* __restrict__ BT,
                                              const float* __restrict__ bias,
                                              bf16_t* __restrict__ C) {
    int lane = threadIdx.x & 31;
    int mt = blockIdx.x, ntc = blockIdx.y;
    int mrow = lane & 15, half = lane >> 4;
    int koff = half ? 8 : 0, klo = half ? 16 : 0;
    const bf16_t* arow = A + ((size_t)mt * 16 + mrow) * HID;
    const bf16_t* brow = BT + ((size_t)ntc * 16 + mrow) * HID;
    v8f acc = {};
#pragma unroll
    for (int kc = 0; kc < 6; ++kc) {
        int k0 = kc * 32;
        v8bf lo = *(const v8bf*)(arow + k0 + klo);
        v8bf hi = *(const v8bf*)(arow + k0 + klo + 8);
        v16bf a;
#pragma unroll
        for (int t = 0; t < 8; ++t) { a[t] = lo[t]; a[t + 8] = hi[t]; }
        v16bf b = *(const v16bf*)(brow + k0 + klo);
        acc = __builtin_amdgcn_wmma_f32_16x16x32_bf16(false, a, false, b, (short)0, acc,
                                                      false, false);
    }
    int col = ntc * 16 + mrow;
    float bv = bias[col];
#pragma unroll
    for (int r = 0; r < 8; ++r)
        C[((size_t)mt * 16 + r + 8 * half) * T1S + col] = (bf16_t)(acc[r] + bv);
}

// ------------------------------------------- WMMA GEMM: root + combine (f32)
__global__ __launch_bounds__(32) void gemm_root(const bf16_t* __restrict__ A,
                                                const bf16_t* __restrict__ BT,
                                                const float* __restrict__ agg,
                                                const float* __restrict__ cb,
                                                float* __restrict__ h, int residual) {
    int lane = threadIdx.x & 31;
    int mt = blockIdx.x, ntc = blockIdx.y;
    int mrow = lane & 15, half = lane >> 4;
    int koff = half ? 8 : 0, klo = half ? 16 : 0;
    const bf16_t* arow = A + ((size_t)mt * 16 + mrow) * HID;
    const bf16_t* brow = BT + ((size_t)ntc * 16 + mrow) * HID;
    v8f acc = {};
#pragma unroll
    for (int kc = 0; kc < 6; ++kc) {
        int k0 = kc * 32;
        v8bf lo = *(const v8bf*)(arow + k0 + klo);
        v8bf hi = *(const v8bf*)(arow + k0 + klo + 8);
        v16bf a;
#pragma unroll
        for (int t = 0; t < 8; ++t) { a[t] = lo[t]; a[t + 8] = hi[t]; }
        v16bf b = *(const v16bf*)(brow + k0 + klo);
        acc = __builtin_amdgcn_wmma_f32_16x16x32_bf16(false, a, false, b, (short)0, acc,
                                                      false, false);
    }
    int col = ntc * 16 + mrow;
    float bv = cb[col];
#pragma unroll
    for (int r = 0; r < 8; ++r) {
        int n = mt * 16 + r + 8 * half;
        size_t idx = (size_t)n * HID + col;
        float v = acc[r] + bv + agg[idx];
        if (residual) v += h[idx];
        h[idx] = v;
    }
}

// ----------------------------------------- BIG kernel: U @ W2T -> scatter agg
// block: 256 thr = 8 waves; tile = 64 edges x 192 cols; grid.y splits j-range.
// B chunks (32k x 192col) staged via async global->LDS through a 4-deep
// circular buffer; prefetch runs 2 stages ahead; one barrier per stage.
__global__ __launch_bounds__(256) void nnconv_msg_scatter(
    const bf16_t* __restrict__ vsrc,   // [E,192]  bf16
    const bf16_t* __restrict__ t1,     // [E,T1S]  bf16 (col192 == 1)
    const bf16_t* __restrict__ w2t,    // [192][KTOT] bf16
    const int* __restrict__ dst,       // [E]
    float* __restrict__ agg)           // [N,192]
{
    __shared__ __align__(16) bf16_t vtile[64 * HID];        // 24.0 KB
    __shared__ __align__(16) bf16_t ttile[64 * JCHUNK];     //  3.0 KB
    __shared__ __align__(16) bf16_t btile[4][HID * BSTR];   // 60.0 KB
    __shared__ int dtile[64];
    const int tid = threadIdx.x;
    const int e0 = blockIdx.x * 64;
    const int jstart = blockIdx.y * JCHUNK;
    const int jend = (jstart + JCHUNK < JTOT) ? (jstart + JCHUNK) : JTOT;
    const int S = (jend - jstart) * 6;

    // ---- async staging of v rows (linear) and t1 j-slices (row segments) ----
    {
        const char* vsg = (const char*)(vsrc + (size_t)e0 * HID);
        for (int i = tid; i < 64 * HID * 2 / 16; i += 256)
            async_g2l_b128((char*)vtile + i * 16, vsg + i * 16);
        for (int i = tid; i < 64 * JCHUNK * 2 / 16; i += 256) {
            int e = i / (JCHUNK / 8);              // 3 chunks per row
            int part = i % (JCHUNK / 8);
            async_g2l_b128(ttile + e * JCHUNK + part * 8,
                           t1 + (size_t)(e0 + e) * T1S + jstart + part * 8);
        }
        if (tid < 64) dtile[tid] = dst[e0 + tid];
    }

    // ---- B-chunk issue: stage s covers K range [ (jstart+s/6)*192 + (s%6)*32 ) ----
    auto issueB = [&](int s) {
        int j = jstart + s / 6, ic = s % 6;
        const bf16_t* gb = w2t + (size_t)j * HID + ic * 32;
        bf16_t* lb = btile[s & 3];
#pragma unroll
        for (int r = 0; r < 3; ++r) {
            int c = tid + r * 256;                 // 0..767
            int col = c >> 2, part = c & 3;
            async_g2l_b128(lb + col * BSTR + part * 8,
                           gb + (size_t)col * KTOT + part * 8);
        }
    };
    issueB(0);
    if (S > 1) issueB(1);

    const int lane = tid & 31;
    const int wv = tid >> 5;
    const int mt = wv >> 1;                    // 0..3 : which 16-edge subtile
    const int nh = wv & 1;                     // 0..1 : which 96-col half
    const int mrow = lane & 15;
    const int half = lane >> 4;
    const int koff = half ? 8 : 0;
    const int klo = half ? 16 : 0;
    const int m_local = mt * 16 + mrow;

    v8f acc[6];
#pragma unroll
    for (int t = 0; t < 6; ++t) acc[t] = (v8f){};

    const bf16_t* vrow = vtile + m_local * HID;
    const bf16_t* trow = ttile + m_local * JCHUNK;

    for (int s = 0; s < S; ++s) {
        if (s + 2 < S)      { issueB(s + 2); wait_async_le6(); }
        else if (s + 1 < S) { wait_async_le3(); }
        else                { wait_async_0(); }
        __syncthreads();                       // stage-s data visible to all waves
        // (buffer s+2 was last read at stage s-2, which completed before the
        //  barrier of iteration s-1 -> safe to overwrite with a single barrier)

        const int jj = s / 6;
        const int i0 = (s % 6) * 32;
        const float tf = (float)trow[jj];      // t1[e, j] broadcast over K
        // build A fragment of U = t1[e,j] * v[e, i0..i0+31]
        v8bf lo = *(const v8bf*)(vrow + i0 + klo);
        v8bf hi = *(const v8bf*)(vrow + i0 + klo + 8);
        v16bf a;
#pragma unroll
        for (int t = 0; t < 8; ++t) {
            a[t]     = (bf16_t)((float)lo[t] * tf);
            a[t + 8] = (bf16_t)((float)hi[t] * tf);
        }
        const bf16_t* bt = btile[s & 3];
#pragma unroll
        for (int nt = 0; nt < 6; ++nt) {
            const int col = nh * 96 + nt * 16 + mrow;
            v8bf blo = *(const v8bf*)(bt + col * BSTR + klo);
            v8bf bhi = *(const v8bf*)(bt + col * BSTR + klo + 8);
            v16bf b;
#pragma unroll
            for (int t = 0; t < 8; ++t) { b[t] = blo[t]; b[t + 8] = bhi[t]; }
            acc[nt] = __builtin_amdgcn_wmma_f32_16x16x32_bf16(
                false, a, false, b, (short)0, acc[nt], false, false);
        }
    }

    // scatter-add to agg[dst[e], col]
#pragma unroll
    for (int nt = 0; nt < 6; ++nt) {
        const int col = nh * 96 + nt * 16 + mrow;
#pragma unroll
        for (int r = 0; r < 8; ++r) {
            const int erow = mt * 16 + r + 8 * half;
            atomicAdd(&agg[(size_t)dtile[erow] * HID + col], acc[nt][r]);
        }
    }
}

// ------------------------------------------------------- LayerNorm + ReLU
__global__ __launch_bounds__(64) void ln_relu_bf16(const float* __restrict__ h,
                                                   const float* __restrict__ g,
                                                   const float* __restrict__ b,
                                                   bf16_t* __restrict__ vbf) {
    int n = blockIdx.x, tid = threadIdx.x;
    __shared__ float red[64];
    const float* row = h + (size_t)n * HID;
    float v0 = row[tid], v1 = row[tid + 64], v2 = row[tid + 128];
    red[tid] = v0 + v1 + v2;
    __syncthreads();
    for (int off = 32; off > 0; off >>= 1) {
        if (tid < off) red[tid] += red[tid + off];
        __syncthreads();
    }
    float mu = red[0] * (1.0f / HID);
    __syncthreads();
    float d0 = v0 - mu, d1 = v1 - mu, d2 = v2 - mu;
    red[tid] = d0 * d0 + d1 * d1 + d2 * d2;
    __syncthreads();
    for (int off = 32; off > 0; off >>= 1) {
        if (tid < off) red[tid] += red[tid + off];
        __syncthreads();
    }
    float rstd = rsqrtf(red[0] * (1.0f / HID) + 1e-5f);
    float z0 = fmaxf(d0 * rstd * g[tid] + b[tid], 0.0f);
    float z1 = fmaxf(d1 * rstd * g[tid + 64] + b[tid + 64], 0.0f);
    float z2 = fmaxf(d2 * rstd * g[tid + 128] + b[tid + 128], 0.0f);
    vbf[(size_t)n * HID + tid] = (bf16_t)z0;
    vbf[(size_t)n * HID + tid + 64] = (bf16_t)z1;
    vbf[(size_t)n * HID + tid + 128] = (bf16_t)z2;
}

// ------------------------------------------------- final LN+ReLU+Linear(2)
__global__ __launch_bounds__(64) void final_out(const float* __restrict__ h,
                                                const float* __restrict__ g,
                                                const float* __restrict__ b,
                                                const float* __restrict__ ow,
                                                const float* __restrict__ ob,
                                                float* __restrict__ out) {
    int n = blockIdx.x, tid = threadIdx.x;
    __shared__ float red[64];
    __shared__ float z[HID];
    const float* row = h + (size_t)n * HID;
    float v0 = row[tid], v1 = row[tid + 64], v2 = row[tid + 128];
    red[tid] = v0 + v1 + v2;
    __syncthreads();
    for (int off = 32; off > 0; off >>= 1) {
        if (tid < off) red[tid] += red[tid + off];
        __syncthreads();
    }
    float mu = red[0] * (1.0f / HID);
    __syncthreads();
    float d0 = v0 - mu, d1 = v1 - mu, d2 = v2 - mu;
    red[tid] = d0 * d0 + d1 * d1 + d2 * d2;
    __syncthreads();
    for (int off = 32; off > 0; off >>= 1) {
        if (tid < off) red[tid] += red[tid + off];
        __syncthreads();
    }
    float rstd = rsqrtf(red[0] * (1.0f / HID) + 1e-5f);
    z[tid] = fmaxf(d0 * rstd * g[tid] + b[tid], 0.0f);
    z[tid + 64] = fmaxf(d1 * rstd * g[tid + 64] + b[tid + 64], 0.0f);
    z[tid + 128] = fmaxf(d2 * rstd * g[tid + 128] + b[tid + 128], 0.0f);
    __syncthreads();
    float p0 = 0.0f, p1 = 0.0f;
    for (int c = tid; c < HID; c += 64) {
        p0 += z[c] * ow[c * 2 + 0];
        p1 += z[c] * ow[c * 2 + 1];
    }
    red[tid] = p0;
    __syncthreads();
    for (int off = 32; off > 0; off >>= 1) {
        if (tid < off) red[tid] += red[tid + off];
        __syncthreads();
    }
    if (tid == 0) out[n * 2 + 0] = red[0] + ob[0];
    __syncthreads();
    red[tid] = p1;
    __syncthreads();
    for (int off = 32; off > 0; off >>= 1) {
        if (tid < off) red[tid] += red[tid + off];
        __syncthreads();
    }
    if (tid == 0) out[n * 2 + 1] = red[0] + ob[1];
}

// =========================================================== host launcher
extern "C" void kernel_launch(void* const* d_in, const int* in_sizes, int n_in,
                              void* d_out, int out_size, void* d_ws, size_t ws_size,
                              hipStream_t stream) {
    const float* x      = (const float*)d_in[0];
    const int*   ei     = (const int*)d_in[1];
    const float* eattr  = (const float*)d_in[2];
    const float* cheb_w = (const float*)d_in[4];
    const float* cheb_b = (const float*)d_in[5];
    const float* eenc_w = (const float*)d_in[6];
    const float* eenc_b = (const float*)d_in[7];
    const float* nn1_w  = (const float*)d_in[8];
    const float* nn1_b  = (const float*)d_in[9];
    const float* nn2_w  = (const float*)d_in[10];
    const float* nn2_b  = (const float*)d_in[11];
    const float* root_w = (const float*)d_in[12];
    const float* conv_b = (const float*)d_in[13];
    const float* ln_g   = (const float*)d_in[14];
    const float* ln_b   = (const float*)d_in[15];
    const float* out_w  = (const float*)d_in[16];
    const float* out_b  = (const float*)d_in[17];
    float* out = (float*)d_out;

    const int* srcI = ei;
    const int* dstI = ei + EE;

    char* wsp = (char*)d_ws;
    auto alloc = [&](size_t bytes) {
        char* p = wsp;
        wsp += (bytes + 255) & ~(size_t)255;
        return p;
    };
    float*  h      = (float*)alloc((size_t)NN * HID * 4);
    float*  agg    = (float*)alloc((size_t)NN * HID * 4);
    bf16_t* vbf    = (bf16_t*)alloc((size_t)NN * HID * 2);
    bf16_t* vsrcB  = (bf16_t*)alloc((size_t)EE * HID * 2);
    bf16_t* eaB    = (bf16_t*)alloc((size_t)EE * HID * 2);
    bf16_t* t1B    = (bf16_t*)alloc((size_t)EE * T1S * 2);
    bf16_t* nn1T   = (bf16_t*)alloc((size_t)NLAY * HID * HID * 2);
    bf16_t* rootT  = (bf16_t*)alloc((size_t)NLAY * HID * HID * 2);
    bf16_t* w2T    = (bf16_t*)alloc((size_t)NLAY * HID * KTOT * 2);
    float*  deg    = (float*)alloc(NN * 4);
    float*  wn     = (float*)alloc(EE * 4);
    float*  txA    = (float*)alloc(NN * CIN * 4);
    float*  txB    = (float*)alloc(NN * CIN * 4);
    float*  txC    = (float*)alloc(NN * CIN * 4);
    float*  tmp    = (float*)alloc(NN * CIN * 4);

    // ---- weight conversions (run every call; deterministic) ----
    conv_t192<<<(NLAY * HID * HID) / 256, 256, 0, stream>>>(nn1_w, nn1T);
    conv_t192<<<(NLAY * HID * HID) / 256, 256, 0, stream>>>(root_w, rootT);
    conv_w2t<<<(NLAY * HID * KTOT) / 256, 256, 0, stream>>>(nn2_w, nn2_b, w2T);
    edge_enc<<<(EE * HID) / 256, 256, 0, stream>>>(eattr, eenc_w, eenc_b, eaB);
    t1_ones<<<EE / 256, 256, 0, stream>>>(t1B);

    // ---- ChebConv ----
    zero_f32<<<(NN + 255) / 256, 256, 0, stream>>>(deg, NN);
    count_deg<<<EE / 256, 256, 0, stream>>>(srcI, deg);
    make_dinv<<<(NN + 255) / 256, 256, 0, stream>>>(deg);
    make_wnorm<<<EE / 256, 256, 0, stream>>>(srcI, dstI, deg, wn);

    cheb_mm<<<(NN * HID) / 256, 256, 0, stream>>>(x, cheb_w, cheb_b, h, 1);
    zero_f32<<<(NN * CIN + 255) / 256, 256, 0, stream>>>(txA, NN * CIN);
    lap_apply<<<EE / 256, 256, 0, stream>>>(x, srcI, dstI, wn, txA);
    cheb_mm<<<(NN * HID) / 256, 256, 0, stream>>>(txA, cheb_w + CIN * HID, cheb_b, h, 0);

    const float* cur0 = x;
    float* cur1 = txA;
    float* nxt = txB;
    float* spare = txC;
    for (int k = 2; k < KCH; ++k) {
        zero_f32<<<(NN * CIN + 255) / 256, 256, 0, stream>>>(tmp, NN * CIN);
        lap_apply<<<EE / 256, 256, 0, stream>>>(cur1, srcI, dstI, wn, tmp);
        cheb_comb<<<(NN * CIN + 255) / 256, 256, 0, stream>>>(tmp, cur0, nxt);
        cheb_mm<<<(NN * HID) / 256, 256, 0, stream>>>(nxt, cheb_w + k * CIN * HID, cheb_b, h, 0);
        float* old1 = cur1;
        cur0 = cur1; cur1 = nxt;
        nxt = (k == 2) ? spare : old1;
    }

    // ---- NNConv layers ----
    f32_to_bf16<<<(NN * HID) / 256, 256, 0, stream>>>(h, vbf, NN * HID);
    for (int L = 0; L < NLAY; ++L) {
        if (L > 0)
            ln_relu_bf16<<<NN, 64, 0, stream>>>(h, ln_g + L * HID, ln_b + L * HID, vbf);
        gather_src<<<(EE * 24) / 256, 256, 0, stream>>>(vbf, srcI, vsrcB);
        gemm_t1<<<dim3(EE / 16, HID / 16), 32, 0, stream>>>(
            eaB, nn1T + (size_t)L * HID * HID, nn1_b + L * HID, t1B);
        zero_f32<<<(NN * HID) / 256, 256, 0, stream>>>(agg, NN * HID);
        nnconv_msg_scatter<<<dim3(EE / 64, KSPLIT), 256, 0, stream>>>(
            vsrcB, t1B, w2T + (size_t)L * HID * KTOT, dstI, agg);
        gemm_root<<<dim3(NN / 16, HID / 16), 32, 0, stream>>>(
            vbf, rootT + (size_t)L * HID * HID, agg, conv_b + L * HID, h, L > 0 ? 1 : 0);
    }

    // ---- final LN + ReLU + Linear(192 -> 2) ----
    final_out<<<NN, 64, 0, stream>>>(h, ln_g, ln_b, out_w, out_b, out);
}
